// ToxAttentiveFP_59219009077540
// MI455X (gfx1250) — matrix operand order repaired and checked by reading, not compiled, not run on hardware
//
#include <hip/hip_runtime.h>

typedef __attribute__((ext_vector_type(16))) __bf16 v16bf;
typedef __attribute__((ext_vector_type(8)))  float  v8f;
typedef __attribute__((ext_vector_type(4)))  float  f32x4;

#define NN_   100000
#define NE_   400000
#define NM_   8192
#define HID_  200
#define H3_   600
#define INCH_ 44
#define EDIM_ 11
#define NT_   12
#define KP_H  224   /* 200 -> pad to mult of 32 */
#define KP_IN 64    /* 44  -> pad */
#define NP_H  256   /* 200 -> pad to mult of 64 (weight rows) */
#define NP_H3 640   /* 600 -> pad */
#define NHEAD 1200
#define NP_HEAD 1216
#define SLOPE_ 0.01f

enum { ACT_NONE = 0, ACT_LEAKY = 1, ACT_RELU = 2, ACT_ELU = 3 };

__device__ __forceinline__ float leakyf(float x) { return x > 0.f ? x : SLOPE_ * x; }
__device__ __forceinline__ float sigmf(float x)  { return 1.f / (1.f + __expf(-x)); }

// ordered-uint encoding so atomicMax(uint) implements float max
__device__ __forceinline__ unsigned encf(float f) {
  unsigned u = __float_as_uint(f);
  return (u >> 31) ? ~u : (u | 0x80000000u);
}
__device__ __forceinline__ float decf(unsigned u) {
  return (u >> 31) ? __uint_as_float(u ^ 0x80000000u) : __uint_as_float(~u);
}
__device__ __forceinline__ void atomAddF(float* p, float v) {
  __hip_atomic_fetch_add(p, v, __ATOMIC_RELAXED, __HIP_MEMORY_SCOPE_AGENT);
}

// ---------------- f32 -> bf16 (RNE) with zero padding ----------------
__global__ void k_tobf(const float* __restrict__ src, unsigned short* __restrict__ dst,
                       int rows, int cols, int srcld, int coloff, int rowsPad, int kpad) {
  int idx = blockIdx.x * blockDim.x + threadIdx.x;
  if (idx >= rowsPad * kpad) return;
  int r = idx / kpad, c = idx - r * kpad;
  float v = (r < rows && c < cols) ? src[(size_t)r * srcld + coloff + c] : 0.f;
  unsigned u = __float_as_uint(v);
  dst[idx] = (unsigned short)((u + 0x7FFFu + ((u >> 16) & 1u)) >> 16);
}

// ---------------- bf16 WMMA GEMM: C[M,Nc] = A[M,Kpad] * W[NcPad,Kpad]^T ------------
// Each wave: 16(M) x 64(N) tile, 4 accumulators, fp32 accumulate.
// KPAD templated -> fully unrolled; sched_barrier enforces a software pipeline:
// fragment loads for k+1 are issued before the WMMA group of k so VMEM latency
// hides under matrix ops (partial s_wait_loadcnt instead of draining to 0).
union FragAB { v16bf bf; f32x4 q[2]; };

template <int KPAD>
__global__ void k_gemm_t(const __bf16* __restrict__ A, const __bf16* __restrict__ W,
                         float* __restrict__ C, const float* __restrict__ bias,
                         int Nc, int NcPad, int act) {
  int lane = threadIdx.x & 31;
  int wv   = threadIdx.x >> 5;
  int n0   = (blockIdx.y * 4 + wv) * 64;
  if (n0 >= NcPad) return;
  int l15 = lane & 15, half = lane >> 4;
  const __bf16* ap = A + ((size_t)blockIdx.x * 16 + l15) * KPAD + half * 16;
  const __bf16* wp = W + (size_t)(n0 + l15) * KPAD + half * 16;

  FragAB a[2];
  FragAB b[2][4];
  v8f acc[4] = {};

  auto loadk = [&](int kk, int s) {
    a[s].q[0] = *(const f32x4*)(ap + kk);       // K = kh..kh+7
    a[s].q[1] = *(const f32x4*)(ap + kk + 8);   // K = kh+8..kh+15
#pragma unroll
    for (int t = 0; t < 4; ++t) {
      const __bf16* w = wp + (size_t)t * 16 * KPAD + kk;
      b[s][t].q[0] = *(const f32x4*)(w);        // K = kh..kh+7
      b[s][t].q[1] = *(const f32x4*)(w + 8);    // K = kh+8..kh+15
    }
  };

  constexpr int NK = KPAD / 32;
  loadk(0, 0);
#pragma unroll
  for (int ik = 0; ik < NK; ++ik) {
    int cur = ik & 1, nxt = cur ^ 1;
    if (ik + 1 < NK) loadk((ik + 1) * 32, nxt);   // prefetch next k-step
    __builtin_amdgcn_sched_barrier(0);            // keep prefetch above this WMMA group
#pragma unroll
    for (int t = 0; t < 4; ++t)
      acc[t] = __builtin_amdgcn_wmma_f32_16x16x32_bf16(
          false, a[cur].bf, false, b[cur][t].bf, (short)0, acc[t], false, false);
  }

  int rbase = blockIdx.x * 16 + half * 8;
#pragma unroll
  for (int t = 0; t < 4; ++t) {
    int col = n0 + t * 16 + l15;
    if (col < Nc) {
      float bv = bias ? bias[col] : 0.f;
#pragma unroll
      for (int v = 0; v < 8; ++v) {
        float xv = acc[t][v] + bv;
        if (act == ACT_LEAKY)      xv = xv > 0.f ? xv : SLOPE_ * xv;
        else if (act == ACT_RELU)  xv = fmaxf(xv, 0.f);
        C[(size_t)(rbase + v) * Nc + col] = xv;
      }
    }
  }
}

// ---------------- per-row dots: y1[r]=X[r]·v1 (opt y2[r]=X[r]·v2) ----------------
__global__ void k_rowdot2(const float* __restrict__ X, int rows, int K,
                          const float* __restrict__ v1, const float* __restrict__ v2,
                          float* __restrict__ y1, float* __restrict__ y2) {
  int w = (blockIdx.x * blockDim.x + threadIdx.x) >> 5;
  int lane = threadIdx.x & 31;
  if (w >= rows) return;
  const float* xr = X + (size_t)w * K;
  float s1 = 0.f, s2 = 0.f;
  for (int c = lane; c < K; c += 32) {
    float xv = xr[c];
    s1 += xv * v1[c];
    if (v2) s2 += xv * v2[c];
  }
  for (int o = 16; o; o >>= 1) {
    s1 += __shfl_xor(s1, o, 32);
    if (v2) s2 += __shfl_xor(s2, o, 32);
  }
  if (lane == 0) { y1[w] = s1; if (v2) y2[w] = s2; }
}

// ---------------- GATEConv edge attention logits (fused edge-feature matmul) -----
// g = leaky(nodepart[j] + edge_attr[e] @ W1b^T); logit = leaky(g·att_l + xattr[i])
__global__ void k_gate_alpha(const int* __restrict__ ei, int E,
                             const float* __restrict__ nodepart,
                             const float* __restrict__ eattr,
                             const float* __restrict__ w1full, /* [200,211] */
                             const float* __restrict__ att_l,
                             const float* __restrict__ xattr,
                             float* __restrict__ logit, unsigned* __restrict__ smax) {
  __shared__ float sW[HID_ * EDIM_];
  __shared__ float sA[HID_];
  for (int t = threadIdx.x; t < HID_ * EDIM_; t += blockDim.x) {
    int r = t / EDIM_, q = t - r * EDIM_;
    sW[t] = w1full[(size_t)r * (HID_ + EDIM_) + HID_ + q];
  }
  for (int t = threadIdx.x; t < HID_; t += blockDim.x) sA[t] = att_l[t];
  __syncthreads();
  int e = (blockIdx.x * blockDim.x + threadIdx.x) >> 5;
  if (e >= E) return;
  int lane = threadIdx.x & 31;
  int j = ei[e], i = ei[E + e];
  float ea[EDIM_];
#pragma unroll
  for (int q = 0; q < EDIM_; ++q) ea[q] = eattr[(size_t)e * EDIM_ + q];
  float part = 0.f;
  for (int c = lane; c < HID_; c += 32) {
    float g = nodepart[(size_t)j * HID_ + c];
#pragma unroll
    for (int q = 0; q < EDIM_; ++q) g += ea[q] * sW[c * EDIM_ + q];
    part += leakyf(g) * sA[c];
  }
  for (int o = 16; o; o >>= 1) part += __shfl_xor(part, o, 32);
  if (lane == 0) {
    float lg = leakyf(part + xattr[i]);
    logit[e] = lg;
    atomicMax(&smax[i], encf(lg));
  }
}

// ---------------- generic segment logit: leaky(asrc[j]+adst[i]) + seg max --------
__global__ void k_seg_logit(const int* __restrict__ jIdx, const int* __restrict__ iIdx,
                            int count, const float* __restrict__ asrc,
                            const float* __restrict__ adst,
                            float* __restrict__ logit, unsigned* __restrict__ smax) {
  int e = blockIdx.x * blockDim.x + threadIdx.x;
  if (e >= count) return;
  int j = jIdx ? jIdx[e] : e;
  int i = iIdx[e];
  float lg = leakyf(asrc[j] + adst[i]);
  logit[e] = lg;
  atomicMax(&smax[i], encf(lg));
}

// ---------------- exp(logit - segmax) + segment sum ----------------
__global__ void k_seg_exp(const float* __restrict__ logit, const int* __restrict__ iIdx,
                          int count, const unsigned* __restrict__ smax,
                          float* __restrict__ ssum, float* __restrict__ eexp) {
  int e = blockIdx.x * blockDim.x + threadIdx.x;
  if (e >= count) return;
  int i = iIdx[e];
  float ex = __expf(logit[e] - decf(smax[i]));
  eexp[e] = ex;
  atomAddF(&ssum[i], ex);
}

// ---------------- normalized scatter: dst[i] += src[j] * (eexp/(ssum[i]+eps)) ----
__global__ void k_scatter(const float* __restrict__ src, const int* __restrict__ jIdx,
                          const int* __restrict__ iIdx, const float* __restrict__ eexp,
                          const float* __restrict__ ssum, float* __restrict__ dst, int count) {
  int e = (blockIdx.x * blockDim.x + threadIdx.x) >> 5;
  if (e >= count) return;
  int lane = threadIdx.x & 31;
  int j = jIdx ? jIdx[e] : e;
  int i = iIdx[e];
  float a = eexp[e] / (ssum[i] + 1e-16f);
  const float* s = src + (size_t)j * HID_;
  float* d = dst + (size_t)i * HID_;
  for (int c = lane; c < HID_; c += 32) atomAddF(&d[c], s[c] * a);
}

// ---------------- plain row scatter-add (initial readout pooling) ----------------
__global__ void k_scatter_rows(const float* __restrict__ src, const int* __restrict__ seg,
                               float* __restrict__ dst, int rows) {
  int n = (blockIdx.x * blockDim.x + threadIdx.x) >> 5;
  if (n >= rows) return;
  int lane = threadIdx.x & 31;
  int i = seg[n];
  const float* s = src + (size_t)n * HID_;
  float* d = dst + (size_t)i * HID_;
  for (int c = lane; c < HID_; c += 32) atomAddF(&d[c], s[c]);
}

// ---------------- y = act(y + bias[c]) ----------------
__global__ void k_bias_act(float* __restrict__ y, const float* __restrict__ bias,
                           int rows, int H, int act) {
  int idx = blockIdx.x * blockDim.x + threadIdx.x;
  if (idx >= rows * H) return;
  int c = idx % H;
  float v = y[idx] + (bias ? bias[c] : 0.f);
  if (act == ACT_ELU)        v = v > 0.f ? v : __expf(v) - 1.f;
  else if (act == ACT_RELU)  v = fmaxf(v, 0.f);
  else if (act == ACT_LEAKY) v = leakyf(v);
  y[idx] = v;
}

// ---------------- GRUCell combine + ReLU, in place on hidden state x ----------------
__global__ void k_gru(const float* __restrict__ gi, const float* __restrict__ gh,
                      float* __restrict__ x, int rows) {
  int idx = blockIdx.x * blockDim.x + threadIdx.x;
  if (idx >= rows * HID_) return;
  int r = idx / HID_, c = idx - r * HID_;
  size_t b = (size_t)r * H3_;
  float rr = sigmf(gi[b + c] + gh[b + c]);
  float zz = sigmf(gi[b + HID_ + c] + gh[b + HID_ + c]);
  float nn = tanhf(gi[b + 2 * HID_ + c] + rr * gh[b + 2 * HID_ + c]);
  size_t xi = (size_t)r * HID_ + c;
  float o = (1.f - zz) * nn + zz * x[xi];
  x[xi] = fmaxf(o, 0.f);
}

// ---------------- per-task head reduction ----------------
__global__ void k_head(const float* __restrict__ h1, const float* __restrict__ w2,
                       const float* __restrict__ b2, float* __restrict__ out, int M) {
  int idx = blockIdx.x * blockDim.x + threadIdx.x;
  if (idx >= M * NT_) return;
  int b = idx / NT_, t = idx - b * NT_;
  const float* hh = h1 + (size_t)b * NHEAD + t * 100;
  const float* ww = w2 + t * 100;
  float s = b2[t];
#pragma unroll 4
  for (int k = 0; k < 100; ++k) s += hh[k] * ww[k];
  out[idx] = s;
}

extern "C" void kernel_launch(void* const* d_in, const int* in_sizes, int n_in,
                              void* d_out, int out_size, void* d_ws, size_t ws_size,
                              hipStream_t stream) {
  (void)in_sizes; (void)n_in; (void)out_size; (void)ws_size;
  const float* x       = (const float*)d_in[0];
  const int*   ei      = (const int*)d_in[1];   // [2,E]: [0..E)=src j, [E..2E)=dst i
  const float* eattr   = (const float*)d_in[2];
  const int*   batch   = (const int*)d_in[3];
  const float* lin1_w  = (const float*)d_in[4];
  const float* lin1_b  = (const float*)d_in[5];
  const float* gw1     = (const float*)d_in[6];   // gate_lin1_w [200,211]
  const float* gw2     = (const float*)d_in[7];   // gate_lin2_w
  const float* g_al    = (const float*)d_in[8];
  const float* g_ar    = (const float*)d_in[9];
  const float* g_b     = (const float*)d_in[10];
  const float* g0wih   = (const float*)d_in[11];
  const float* g0whh   = (const float*)d_in[12];
  const float* g0bih   = (const float*)d_in[13];
  const float* g0bhh   = (const float*)d_in[14];
  const float* cw      = (const float*)d_in[15];  // [5,200,200]
  const float* c_as    = (const float*)d_in[16];  // [5,200]
  const float* c_ad    = (const float*)d_in[17];
  const float* c_b     = (const float*)d_in[18];
  const float* gwih    = (const float*)d_in[19];  // [5,600,200]
  const float* gwhh    = (const float*)d_in[20];
  const float* gbih    = (const float*)d_in[21];
  const float* gbhh    = (const float*)d_in[22];
  const float* mw      = (const float*)d_in[23];
  const float* m_as    = (const float*)d_in[24];
  const float* m_ad    = (const float*)d_in[25];
  const float* m_b     = (const float*)d_in[26];
  const float* mgwih   = (const float*)d_in[27];
  const float* mgwhh   = (const float*)d_in[28];
  const float* mgbih   = (const float*)d_in[29];
  const float* mgbhh   = (const float*)d_in[30];
  const float* lin2_w  = (const float*)d_in[31];
  const float* lin2_b  = (const float*)d_in[32];
  const float* hw1     = (const float*)d_in[33];  // [12,100,200] = [1200,200]
  const float* hb1     = (const float*)d_in[34];  // [1200]
  const float* hw2     = (const float*)d_in[35];  // [12,100]
  const float* hb2     = (const float*)d_in[36];
  float* out = (float*)d_out;

  // ---- bump allocator over d_ws ----
  char* pp = (char*)d_ws;
  auto alloc = [&](size_t bytes) -> void* {
    void* r = (void*)pp;
    pp += (bytes + 255) & ~(size_t)255;
    return r;
  };
  // bf16 weight staging
  void* wb_lin1  = alloc((size_t)NP_H * KP_IN * 2);
  void* wb_w1a   = alloc((size_t)NP_H * KP_H * 2);
  void* wb_gw2   = alloc((size_t)NP_H * KP_H * 2);
  void* wb_g0wih = alloc((size_t)NP_H3 * KP_H * 2);
  void* wb_g0whh = alloc((size_t)NP_H3 * KP_H * 2);
  void* wb_cw[5], *wb_gwih[5], *wb_gwhh[5];
  for (int l = 0; l < 5; ++l) wb_cw[l]   = alloc((size_t)NP_H * KP_H * 2);
  for (int l = 0; l < 5; ++l) wb_gwih[l] = alloc((size_t)NP_H3 * KP_H * 2);
  for (int l = 0; l < 5; ++l) wb_gwhh[l] = alloc((size_t)NP_H3 * KP_H * 2);
  void* wb_mw    = alloc((size_t)NP_H * KP_H * 2);
  void* wb_mgwih = alloc((size_t)NP_H3 * KP_H * 2);
  void* wb_mgwhh = alloc((size_t)NP_H3 * KP_H * 2);
  void* wb_lin2  = alloc((size_t)NP_H * KP_H * 2);
  void* wb_hw1   = alloc((size_t)NP_HEAD * KP_H * 2);
  // activations
  float* xn   = (float*)alloc((size_t)NN_ * HID_ * 4);
  float* hb   = (float*)alloc((size_t)NN_ * HID_ * 4);
  float* gi   = (float*)alloc((size_t)NN_ * H3_ * 4);   // also nodepart [N,200]
  float* gh   = (float*)alloc((size_t)NN_ * H3_ * 4);   // also t / xp  [N,200]
  float* xs   = (float*)alloc((size_t)NN_ * HID_ * 4);
  void*  xnbf = alloc((size_t)NN_ * KP_H * 2);
  void*  hbbf = alloc((size_t)NN_ * KP_H * 2);
  void*  obf  = alloc((size_t)NM_ * KP_H * 2);
  void*  ebf  = alloc((size_t)NM_ * KP_H * 2);
  float* dot1 = (float*)alloc((size_t)NN_ * 4);
  float* dot2 = (float*)alloc((size_t)NN_ * 4);
  float* elog = (float*)alloc((size_t)NE_ * 4);
  float* eexp = (float*)alloc((size_t)NE_ * 4);
  unsigned* smax = (unsigned*)alloc((size_t)NN_ * 4);
  float* ssum = (float*)alloc((size_t)NN_ * 4);
  float* mout = (float*)alloc((size_t)NM_ * HID_ * 4);
  float* moutw= (float*)alloc((size_t)NM_ * HID_ * 4);
  float* emb  = (float*)alloc((size_t)NM_ * HID_ * 4);
  float* h1   = (float*)alloc((size_t)NM_ * NHEAD * 4);

  auto tobf = [&](const float* src, void* dst, int rows, int cols, int srcld, int coloff,
                  int rowsPad, int kpad) {
    int tot = rowsPad * kpad;
    k_tobf<<<(tot + 255) / 256, 256, 0, stream>>>(src, (unsigned short*)dst, rows, cols,
                                                  srcld, coloff, rowsPad, kpad);
  };
  auto gemm = [&](const void* A, const void* W, float* C, const float* bias,
                  int Mr, int Nc, int NcPad, int Kpad, int act) {
    dim3 g(Mr / 16, (NcPad + 255) / 256);
    if (Kpad == KP_H)
      k_gemm_t<KP_H><<<g, 128, 0, stream>>>((const __bf16*)A, (const __bf16*)W, C, bias,
                                            Nc, NcPad, act);
    else
      k_gemm_t<KP_IN><<<g, 128, 0, stream>>>((const __bf16*)A, (const __bf16*)W, C, bias,
                                             Nc, NcPad, act);
  };
  auto zero = [&](void* ptr, size_t bytes) { hipMemsetAsync(ptr, 0, bytes, stream); };

  // ---- stage weights in bf16 (padded) ----
  tobf(lin1_w, wb_lin1, 200, 44, 44, 0, NP_H, KP_IN);
  tobf(gw1, wb_w1a, 200, 200, 211, 0, NP_H, KP_H);          // W1a = cols 0..199
  tobf(gw2, wb_gw2, 200, 200, 200, 0, NP_H, KP_H);
  tobf(g0wih, wb_g0wih, 600, 200, 200, 0, NP_H3, KP_H);
  tobf(g0whh, wb_g0whh, 600, 200, 200, 0, NP_H3, KP_H);
  for (int l = 0; l < 5; ++l) {
    tobf(cw + (size_t)l * 200 * 200, wb_cw[l], 200, 200, 200, 0, NP_H, KP_H);
    tobf(gwih + (size_t)l * 600 * 200, wb_gwih[l], 600, 200, 200, 0, NP_H3, KP_H);
    tobf(gwhh + (size_t)l * 600 * 200, wb_gwhh[l], 600, 200, 200, 0, NP_H3, KP_H);
  }
  tobf(mw, wb_mw, 200, 200, 200, 0, NP_H, KP_H);
  tobf(mgwih, wb_mgwih, 600, 200, 200, 0, NP_H3, KP_H);
  tobf(mgwhh, wb_mgwhh, 600, 200, 200, 0, NP_H3, KP_H);
  tobf(lin2_w, wb_lin2, 200, 200, 200, 0, NP_H, KP_H);
  tobf(hw1, wb_hw1, NHEAD, 200, 200, 0, NP_HEAD, KP_H);

  const int* ej = ei;        // source nodes j
  const int* eiT = ei + NE_; // target nodes i

  // ---- lin1: xn = leaky(x @ lin1_w^T + b) ----
  tobf(x, xnbf, NN_, 44, 44, 0, NN_, KP_IN);
  gemm(xnbf, wb_lin1, xn, lin1_b, NN_, HID_, NP_H, KP_IN, ACT_LEAKY);
  tobf(xn, xnbf, NN_, HID_, HID_, 0, NN_, KP_H);

  // ---- GATEConv ----
  gemm(xnbf, wb_w1a, gi, nullptr, NN_, HID_, NP_H, KP_H, ACT_NONE);   // nodepart
  gemm(xnbf, wb_gw2, gh, nullptr, NN_, HID_, NP_H, KP_H, ACT_NONE);   // t = xn@gate_lin2
  k_rowdot2<<<(NN_ * 32 + 255) / 256, 256, 0, stream>>>(xn, NN_, HID_, g_ar, nullptr, dot1, nullptr);
  zero(smax, (size_t)NN_ * 4); zero(ssum, (size_t)NN_ * 4);
  k_gate_alpha<<<(NE_ * 32 + 255) / 256, 256, 0, stream>>>(ei, NE_, gi, eattr, gw1, g_al,
                                                           dot1, elog, smax);
  k_seg_exp<<<(NE_ + 255) / 256, 256, 0, stream>>>(elog, eiT, NE_, smax, ssum, eexp);
  zero(hb, (size_t)NN_ * HID_ * 4);
  k_scatter<<<(NE_ * 32 + 255) / 256, 256, 0, stream>>>(gh, ej, eiT, eexp, ssum, hb, NE_);
  k_bias_act<<<(NN_ * HID_ + 255) / 256, 256, 0, stream>>>(hb, g_b, NN_, HID_, ACT_ELU);
  tobf(hb, hbbf, NN_, HID_, HID_, 0, NN_, KP_H);
  gemm(hbbf, wb_g0wih, gi, g0bih, NN_, H3_, NP_H3, KP_H, ACT_NONE);
  gemm(xnbf, wb_g0whh, gh, g0bhh, NN_, H3_, NP_H3, KP_H, ACT_NONE);
  k_gru<<<(NN_ * HID_ + 255) / 256, 256, 0, stream>>>(gi, gh, xn, NN_);
  tobf(xn, xnbf, NN_, HID_, HID_, 0, NN_, KP_H);

  // ---- 5 GATConv layers ----
  for (int l = 0; l < 5; ++l) {
    gemm(xnbf, wb_cw[l], gh, nullptr, NN_, HID_, NP_H, KP_H, ACT_NONE);   // xp in gh
    k_rowdot2<<<(NN_ * 32 + 255) / 256, 256, 0, stream>>>(gh, NN_, HID_, c_as + l * HID_,
                                                          c_ad + l * HID_, dot1, dot2);
    zero(smax, (size_t)NN_ * 4); zero(ssum, (size_t)NN_ * 4);
    k_seg_logit<<<(NE_ + 255) / 256, 256, 0, stream>>>(ej, eiT, NE_, dot1, dot2, elog, smax);
    k_seg_exp<<<(NE_ + 255) / 256, 256, 0, stream>>>(elog, eiT, NE_, smax, ssum, eexp);
    zero(hb, (size_t)NN_ * HID_ * 4);
    k_scatter<<<(NE_ * 32 + 255) / 256, 256, 0, stream>>>(gh, ej, eiT, eexp, ssum, hb, NE_);
    k_bias_act<<<(NN_ * HID_ + 255) / 256, 256, 0, stream>>>(hb, c_b + l * HID_, NN_, HID_, ACT_ELU);
    tobf(hb, hbbf, NN_, HID_, HID_, 0, NN_, KP_H);
    gemm(hbbf, wb_gwih[l], gi, gbih + l * H3_, NN_, H3_, NP_H3, KP_H, ACT_NONE);
    gemm(xnbf, wb_gwhh[l], gh, gbhh + l * H3_, NN_, H3_, NP_H3, KP_H, ACT_NONE);
    k_gru<<<(NN_ * HID_ + 255) / 256, 256, 0, stream>>>(gi, gh, xn, NN_);
    tobf(xn, xnbf, NN_, HID_, HID_, 0, NN_, KP_H);
  }

  // ---- molecule-level attentive readout ----
  zero(mout, (size_t)NM_ * HID_ * 4);
  k_scatter_rows<<<(NN_ * 32 + 255) / 256, 256, 0, stream>>>(xn, batch, mout, NN_);
  k_bias_act<<<(NM_ * HID_ + 255) / 256, 256, 0, stream>>>(mout, nullptr, NM_, HID_, ACT_RELU);
  gemm(xnbf, wb_mw, xs, nullptr, NN_, HID_, NP_H, KP_H, ACT_NONE);      // xs = xn@mol_w
  k_rowdot2<<<(NN_ * 32 + 255) / 256, 256, 0, stream>>>(xs, NN_, HID_, m_as, nullptr, dot1, nullptr);
  for (int ts = 0; ts < 2; ++ts) {
    tobf(mout, obf, NM_, HID_, HID_, 0, NM_, KP_H);
    gemm(obf, wb_mw, moutw, nullptr, NM_, HID_, NP_H, KP_H, ACT_NONE);  // out@mol_w
    k_rowdot2<<<(NM_ * 32 + 255) / 256, 256, 0, stream>>>(moutw, NM_, HID_, m_ad, nullptr, dot2, nullptr);
    zero(smax, (size_t)NM_ * 4); zero(ssum, (size_t)NM_ * 4);
    k_seg_logit<<<(NN_ + 255) / 256, 256, 0, stream>>>(nullptr, batch, NN_, dot1, dot2, elog, smax);
    k_seg_exp<<<(NN_ + 255) / 256, 256, 0, stream>>>(elog, batch, NN_, smax, ssum, eexp);
    zero(hb, (size_t)NM_ * HID_ * 4);
    k_scatter<<<(NN_ * 32 + 255) / 256, 256, 0, stream>>>(xs, nullptr, batch, eexp, ssum, hb, NN_);
    k_bias_act<<<(NM_ * HID_ + 255) / 256, 256, 0, stream>>>(hb, m_b, NM_, HID_, ACT_ELU);
    tobf(hb, hbbf, NM_, HID_, HID_, 0, NM_, KP_H);
    gemm(hbbf, wb_mgwih, gi, mgbih, NM_, H3_, NP_H3, KP_H, ACT_NONE);
    gemm(obf, wb_mgwhh, gh, mgbhh, NM_, H3_, NP_H3, KP_H, ACT_NONE);
    k_gru<<<(NM_ * HID_ + 255) / 256, 256, 0, stream>>>(gi, gh, mout, NM_);
  }

  // ---- final linear + per-task heads ----
  tobf(mout, obf, NM_, HID_, HID_, 0, NM_, KP_H);
  gemm(obf, wb_lin2, emb, lin2_b, NM_, HID_, NP_H, KP_H, ACT_NONE);
  tobf(emb, ebf, NM_, HID_, HID_, 0, NM_, KP_H);
  gemm(ebf, wb_hw1, h1, hb1, NM_, NHEAD, NP_HEAD, KP_H, ACT_RELU);
  k_head<<<(NM_ * NT_ + 255) / 256, 256, 0, stream>>>(h1, hw2, hb2, out, NM_);
}
